// MyLSTM_23398981828704
// MI455X (gfx1250) — matrix operand order repaired and checked by reading, not compiled, not run on hardware
//
#include <hip/hip_runtime.h>
#include <hip/hip_bf16.h>

#define NDIM 768
#define SEQ  256

typedef __attribute__((ext_vector_type(8)))  _Float16 h8;
typedef __attribute__((ext_vector_type(16))) _Float16 h16;
typedef __attribute__((ext_vector_type(8)))  float    f8;

__device__ __forceinline__ h16 cat16(h8 lo, h8 hi) {
  return __builtin_shufflevector(lo, hi, 0,1,2,3,4,5,6,7,8,9,10,11,12,13,14,15);
}

// A-fragment (16x32 f16, row-major, ld=NDIM): lane holds K {0..15} / {16..31}
__device__ __forceinline__ h16 ldfragA(const _Float16* p) {
  return cat16(*(const h8*)p, *(const h8*)(p + 8));
}
// B-fragment (32x16 f16) from transposed matrix BT[n][k]: 16 contiguous K per lane
__device__ __forceinline__ h16 ldfragB(const _Float16* p) {
  return cat16(*(const h8*)p, *(const h8*)(p + 8));
}

#define WMMA_F16(a, b, c) \
  __builtin_amdgcn_wmma_f32_16x16x32_f16(false, (a), false, (b), (short)0, (c), false, false)

__device__ __forceinline__ float sigmoidf_(float x) { return 1.0f / (1.0f + __expf(-x)); }
__device__ __forceinline__ float tanhf_(float x)    { return 1.0f - 2.0f / (__expf(2.0f * x) + 1.0f); }

// ---------------- prep kernels ----------------

__global__ void convert_f16_kernel(const float* __restrict__ src, _Float16* __restrict__ dst, int n) {
  int i = blockIdx.x * blockDim.x + threadIdx.x;
  if (i < n) dst[i] = (_Float16)src[i];
}

__global__ void bias_sum_kernel(const float* __restrict__ a, const float* __restrict__ b,
                                float* __restrict__ dst, int n) {
  int i = blockIdx.x * blockDim.x + threadIdx.x;
  if (i < n) dst[i] = a[i] + b[i];
}

__global__ void zero_f16_kernel(_Float16* __restrict__ p, int n) {
  int i = blockIdx.x * blockDim.x + threadIdx.x;
  if (i < n) p[i] = (_Float16)0.0f;
}

__global__ void zero_f32_kernel(float* __restrict__ p, int n) {
  int i = blockIdx.x * blockDim.x + threadIdx.x;
  if (i < n) p[i] = 0.0f;
}

// Xbf[t][n][k] = f16( x[k][t][n] ); 64x64 tile transpose through LDS per (t, tile)
__global__ __launch_bounds__(256) void transpose_x_kernel(const float* __restrict__ x,
                                                          _Float16* __restrict__ Xbf) {
  __shared__ _Float16 tile[64][66];
  const int t = blockIdx.z;
  const int kBase = blockIdx.x * 64;
  const int nBase = blockIdx.y * 64;
  const int c  = threadIdx.x & 63;
  const int r0 = threadIdx.x >> 6;
#pragma unroll
  for (int rr = 0; rr < 16; ++rr) {
    int kk = r0 + rr * 4;
    float v = x[(size_t)(kBase + kk) * (SEQ * NDIM) + (size_t)t * NDIM + nBase + c];
    tile[kk][c] = (_Float16)v;
  }
  __syncthreads();
#pragma unroll
  for (int rr = 0; rr < 16; ++rr) {
    int nn = r0 + rr * 4;
    Xbf[(size_t)t * NDIM * NDIM + (size_t)(nBase + nn) * NDIM + kBase + c] = tile[c][nn];
  }
}

// ---------------- fused LSTM step ----------------
// Grid (6,12); 256 threads = 8 waves (4m x 2n); wave tile 32x32, all 4 gates in registers.
// 8 weight-matrix stages per K-chunk (U_i,V_i,U_f,V_f,U_o,V_o,U_c,V_c).
// A-fragments: 4 rotating buffers, prefetch distance 2, continuous across K-chunks
// (stages 6/7 load next-chunk stages 0/1).  B-fragments double-buffered across K-chunks.

#define NN_ ((size_t)NDIM * NDIM)

// Stage J: load A for stage J+2 into (T0,T1) from Mbase[MATIDX] at kb+KOFF,
// (at J==4) issue next-K B loads, then 4 WMMAs using (U0,U1).
// Even J -> U_g x X fragments, odd J -> V_g x h fragments.
#define STAGE(J, U0, U1, T0, T1, MATIDX, KOFF)                              \
  do {                                                                      \
    T0 = ldfragA(Mbase[MATIDX] + aoff0 + kb + (KOFF));                      \
    T1 = ldfragA(Mbase[MATIDX] + aoff1 + kb + (KOFF));                      \
    if ((J) == 4) {                                                         \
      nbx0 = ldfragB(Xt + boff0 + kb + 32);                                 \
      nbx1 = ldfragB(Xt + boff1 + kb + 32);                                 \
      nbh0 = ldfragB(hT_in + boff0 + kb + 32);                              \
      nbh1 = ldfragB(hT_in + boff1 + kb + 32);                              \
    }                                                                       \
    {                                                                       \
      const int g_ = (J) >> 1;                                              \
      if ((J) & 1) {                                                        \
        acc[g_][0][0] = WMMA_F16(U0, bh0, acc[g_][0][0]);                   \
        acc[g_][0][1] = WMMA_F16(U0, bh1, acc[g_][0][1]);                   \
        acc[g_][1][0] = WMMA_F16(U1, bh0, acc[g_][1][0]);                   \
        acc[g_][1][1] = WMMA_F16(U1, bh1, acc[g_][1][1]);                   \
      } else {                                                              \
        acc[g_][0][0] = WMMA_F16(U0, bx0, acc[g_][0][0]);                   \
        acc[g_][0][1] = WMMA_F16(U0, bx1, acc[g_][0][1]);                   \
        acc[g_][1][0] = WMMA_F16(U1, bx0, acc[g_][1][0]);                   \
        acc[g_][1][1] = WMMA_F16(U1, bx1, acc[g_][1][1]);                   \
      }                                                                     \
    }                                                                       \
  } while (0)

__global__ __launch_bounds__(256) void lstm_step_kernel(
    const _Float16* __restrict__ Wbf,    // [8][N][N]: U_i..U_c, V_i..V_c (f16 row-major)
    const _Float16* __restrict__ Xbf,    // [SEQ][n][k] transposed f16 x
    const float*    __restrict__ bsum,   // [4][N]
    const _Float16* __restrict__ hT_in,  // [n][m] f16
    _Float16*       __restrict__ hT_out, // [n][m] f16
    float*          __restrict__ cbuf,   // [m][n] f32
    float*          __restrict__ out_hidden, // [m][t][n] f32
    float*          __restrict__ out_tail,   // h_t then c_t
    int t)
{
  __shared__ _Float16 hlds[8][32][40];   // per-wave transpose staging (pitch 40 -> 16B rows)

  const int lane  = threadIdx.x & 31;
  const int wave  = threadIdx.x >> 5;
  const int mW = blockIdx.x * 128 + (wave >> 1) * 32;
  const int nW = blockIdx.y * 64  + (wave & 1) * 32;
  const int l15  = lane & 15;
  const int half = lane >> 4;

  const size_t aoff0 = (size_t)(mW + l15) * NDIM + half * 16;
  const size_t aoff1 = aoff0 + (size_t)16 * NDIM;
  const size_t boff0 = (size_t)(nW + l15) * NDIM + half * 16;
  const size_t boff1 = boff0 + (size_t)16 * NDIM;

  const _Float16* Xt = Xbf + (size_t)t * NN_;

  // Warm L2 with next timestep's X slab: 576 waves x 32 lanes x 64B == slab size exactly.
  if (t + 1 < SEQ) {
    const size_t chunk = ((size_t)(blockIdx.y * gridDim.x + blockIdx.x) * 8 + wave);
    const char* pf = (const char*)(Xt + NN_) + chunk * 2048 + (size_t)lane * 64;
    __builtin_prefetch(pf, 0, 2);
  }

  // Stage order: U_i, V_i, U_f, V_f, U_o, V_o, U_c, V_c
  const _Float16* Mbase[8] = {
    Wbf + 0 * NN_, Wbf + 4 * NN_,
    Wbf + 1 * NN_, Wbf + 5 * NN_,
    Wbf + 2 * NN_, Wbf + 6 * NN_,
    Wbf + 3 * NN_, Wbf + 7 * NN_ };

  f8 acc[4][2][2] = {};   // [gate][tm][tn]

  h16 bx0 = ldfragB(Xt + boff0);
  h16 bx1 = ldfragB(Xt + boff1);
  h16 bh0 = ldfragB(hT_in + boff0);
  h16 bh1 = ldfragB(hT_in + boff1);

  // Prime the A pipeline: stages 0 and 1 of kb = 0.
  h16 P0a = ldfragA(Mbase[0] + aoff0);
  h16 P0b = ldfragA(Mbase[0] + aoff1);
  h16 P1a = ldfragA(Mbase[1] + aoff0);
  h16 P1b = ldfragA(Mbase[1] + aoff1);
  h16 P2a, P2b, P3a, P3b;

  for (int kb = 0; kb < NDIM; kb += 32) {
    h16 nbx0, nbx1, nbh0, nbh1;
    // stage j uses buf[j%4], loads stage j+2 into buf[(j+2)%4]; 8 % 4 == 0 so the
    // rotation is loop-invariant; stages 6/7 fetch next-chunk stages 0/1 (kb+32).
    STAGE(0, P0a, P0b, P2a, P2b, 2, 0);
    STAGE(1, P1a, P1b, P3a, P3b, 3, 0);
    STAGE(2, P2a, P2b, P0a, P0b, 4, 0);
    STAGE(3, P3a, P3b, P1a, P1b, 5, 0);
    STAGE(4, P0a, P0b, P2a, P2b, 6, 0);
    STAGE(5, P1a, P1b, P3a, P3b, 7, 0);
    STAGE(6, P2a, P2b, P0a, P0b, 0, 32);
    STAGE(7, P3a, P3b, P1a, P1b, 1, 32);
    bx0 = nbx0; bx1 = nbx1; bh0 = nbh0; bh1 = nbh1;   // rotate B buffers
  }

  const int nLane = l15;
  const bool last = (t == SEQ - 1);
#pragma unroll
  for (int tm = 0; tm < 2; ++tm) {
#pragma unroll
    for (int r = 0; r < 8; ++r) {
      const int ml = tm * 16 + r + half * 8;  // local m in 0..31
      const int m  = mW + ml;
      const float bi = bsum[0 * NDIM + m];
      const float bf = bsum[1 * NDIM + m];
      const float bo = bsum[2 * NDIM + m];
      const float bc = bsum[3 * NDIM + m];
#pragma unroll
      for (int tn = 0; tn < 2; ++tn) {
        const int nl = tn * 16 + nLane;       // local n in 0..31
        const int n  = nW + nl;
        float zi = acc[0][tm][tn][r] + bi;
        float zf = acc[1][tm][tn][r] + bf;
        float zo = acc[2][tm][tn][r] + bo;
        float zg = acc[3][tm][tn][r] + bc;
        float ig = sigmoidf_(zi);
        float fg = sigmoidf_(zf);
        float og = sigmoidf_(zo);
        float gg = tanhf_(zg);
        const size_t cm = (size_t)m * NDIM + n;
        float cn = fg * cbuf[cm] + ig * gg;
        float hn = og * tanhf_(cn);
        cbuf[cm] = cn;
        out_hidden[((size_t)m * SEQ + t) * NDIM + n] = hn;
        hlds[wave][nl][ml] = (_Float16)hn;    // stage transposed tile in LDS
        if (last) {
          out_tail[cm] = hn;
          out_tail[NN_ + cm] = cn;
        }
      }
    }
  }

  // Coalesced hT_out store: lane owns row n = nW + lane, 32 f16 = 4 x b128.
  {
    const _Float16* src = &hlds[wave][lane][0];
    h8 r0 = *(const h8*)(src);
    h8 r1 = *(const h8*)(src + 8);
    h8 r2 = *(const h8*)(src + 16);
    h8 r3 = *(const h8*)(src + 24);
    _Float16* dst = hT_out + (size_t)(nW + lane) * NDIM + mW;
    *(h8*)(dst)      = r0;
    *(h8*)(dst + 8)  = r1;
    *(h8*)(dst + 16) = r2;
    *(h8*)(dst + 24) = r3;
  }
}

// ---------------- host launcher ----------------

extern "C" void kernel_launch(void* const* d_in, const int* in_sizes, int n_in,
                              void* d_out, int out_size, void* d_ws, size_t ws_size,
                              hipStream_t stream) {
  (void)in_sizes; (void)n_in; (void)out_size; (void)ws_size;
  constexpr size_t NN = (size_t)NDIM * NDIM;

  const float* x = (const float*)d_in[0];
  const float* U[4], *V[4], *BU[4], *BV[4];
  for (int g = 0; g < 4; ++g) {
    U[g]  = (const float*)d_in[1 + 4 * g];
    V[g]  = (const float*)d_in[2 + 4 * g];
    BU[g] = (const float*)d_in[3 + 4 * g];
    BV[g] = (const float*)d_in[4 + 4 * g];
  }

  char* ws = (char*)d_ws;
  _Float16* Xbf  = (_Float16*)ws;                                    // 301,989,888 B
  _Float16* Wbf  = (_Float16*)(ws + (size_t)SEQ * NN * 2);           //   9,437,184 B
  float*    bsum = (float*)(ws + (size_t)SEQ * NN * 2 + 8 * NN * 2); //      12,288 B
  _Float16* hT0  = (_Float16*)((char*)bsum + 4 * NDIM * sizeof(float));
  _Float16* hT1  = hT0 + NN;
  float*    cbuf = (float*)((char*)(hT1 + NN));
  // total ws usage ~316 MB

  const int cvtBlocks = (int)((NN + 255) / 256);
  for (int g = 0; g < 4; ++g) {
    convert_f16_kernel<<<cvtBlocks, 256, 0, stream>>>(U[g], Wbf + (size_t)g * NN, (int)NN);
    convert_f16_kernel<<<cvtBlocks, 256, 0, stream>>>(V[g], Wbf + (size_t)(4 + g) * NN, (int)NN);
    bias_sum_kernel<<<3, 256, 0, stream>>>(BU[g], BV[g], bsum + g * NDIM, NDIM);
  }
  zero_f16_kernel<<<cvtBlocks, 256, 0, stream>>>(hT0, (int)NN);
  zero_f16_kernel<<<cvtBlocks, 256, 0, stream>>>(hT1, (int)NN);
  zero_f32_kernel<<<cvtBlocks, 256, 0, stream>>>(cbuf, (int)NN);

  transpose_x_kernel<<<dim3(NDIM / 64, NDIM / 64, SEQ), 256, 0, stream>>>(x, Xbf);

  float* out      = (float*)d_out;
  float* out_tail = out + (size_t)NDIM * SEQ * NDIM;

  dim3 grid(NDIM / 128, NDIM / 64, 1);
  for (int t = 0; t < SEQ; ++t) {
    const _Float16* hin  = (t & 1) ? hT1 : hT0;
    _Float16*       hout = (t & 1) ? hT0 : hT1;
    lstm_step_kernel<<<grid, 256, 0, stream>>>(Wbf, Xbf, bsum, hin, hout, cbuf,
                                               out, out_tail, t);
  }
}
